// Block_46643344834722
// MI455X (gfx1250) — matrix operand order repaired and checked by reading, not compiled, not run on hardware
//
#include <hip/hip_runtime.h>
#include <hip/hip_bf16.h>

// MI455X / gfx1250, wave32. GEMMs run on v_wmma_f32_16x16x32_f16 with fp32
// accumulate; fp32 weights are converted to f16 in-register while staging
// through a ping-pong LDS buffer (register-prefetched one K-step ahead) so
// the HBM weight stream (~201 MB, the roofline floor at 23.3 TB/s) overlaps
// with the WMMA work. Weight staging stores one ds_store_b128 per thread.

typedef __attribute__((ext_vector_type(16))) _Float16 v16h;
typedef __attribute__((ext_vector_type(8)))  _Float16 v8h;
typedef __attribute__((ext_vector_type(8)))  float    v8f;

#define D_MODEL 2048
#define NHEAD   16
#define DHEAD   128
#define BATCH   512
#define DFF     8192
#define LN_EPS  1e-5f

#if __has_builtin(__builtin_amdgcn_global_load_async_to_lds_b32) && \
    __has_builtin(__builtin_amdgcn_s_wait_asynccnt)
#define USE_ASYNC_LDS 1
typedef __attribute__((address_space(1))) int as1_int;   // global
typedef __attribute__((address_space(3))) int as3_int;   // LDS
#else
#define USE_ASYNC_LDS 0
#endif

// ------------------------------------------------------------------
// LayerNorm: fp32 row in -> f16 row out (feeds WMMA A operand)
// ------------------------------------------------------------------
__global__ void __launch_bounds__(256)
ln_f16_kernel(const float* __restrict__ x, const float* __restrict__ g,
              const float* __restrict__ b, _Float16* __restrict__ out)
{
    __shared__ float red[256];
    const int row = blockIdx.x;
    const int tid = threadIdx.x;
    const float* xr = x + (size_t)row * D_MODEL;

    float vals[8];
    float s = 0.f;
#pragma unroll
    for (int i = 0; i < 8; ++i) { vals[i] = xr[tid + i * 256]; s += vals[i]; }
    red[tid] = s; __syncthreads();
    for (int off = 128; off > 0; off >>= 1) {
        if (tid < off) red[tid] += red[tid + off];
        __syncthreads();
    }
    const float mean = red[0] * (1.f / D_MODEL);
    __syncthreads();

    float v = 0.f;
#pragma unroll
    for (int i = 0; i < 8; ++i) { float d = vals[i] - mean; v += d * d; }
    red[tid] = v; __syncthreads();
    for (int off = 128; off > 0; off >>= 1) {
        if (tid < off) red[tid] += red[tid + off];
        __syncthreads();
    }
    const float rstd = rsqrtf(red[0] * (1.f / D_MODEL) + LN_EPS);

    _Float16* orow = out + (size_t)row * D_MODEL;
#pragma unroll
    for (int i = 0; i < 8; ++i) {
        const int c = tid + i * 256;
        orow[c] = (_Float16)((vals[i] - mean) * rstd * g[c] + b[c]);
    }
}

// ------------------------------------------------------------------
// Per-sample outer-product attention: one block per (b,h), 128 thr.
// attn[i,j] = softmax_j(q_i * k_j / sqrt(DH)); o_i = sum_j attn * v_j
// k/v staging is a pure global->LDS copy: uses the gfx1250 async-LDS
// path (GLOBAL_LOAD_ASYNC_TO_LDS_B32 + s_wait_asynccnt) when available.
// ------------------------------------------------------------------
__global__ void __launch_bounds__(128)
attn_kernel(const float* __restrict__ q, const float* __restrict__ k,
            const float* __restrict__ v, _Float16* __restrict__ o)
{
    __shared__ float ks[DHEAD], vs[DHEAD];
    const int bh = blockIdx.x;           // b*NHEAD + h
    const int t  = threadIdx.x;          // i index (row of attn)
    const size_t base = (size_t)bh * DHEAD;

#if USE_ASYNC_LDS
    __builtin_amdgcn_global_load_async_to_lds_b32(
        (as1_int*)(k + base + t), (as3_int*)(&ks[t]), 0, 0);
    __builtin_amdgcn_global_load_async_to_lds_b32(
        (as1_int*)(v + base + t), (as3_int*)(&vs[t]), 0, 0);
    __builtin_amdgcn_s_wait_asynccnt(0);
#else
    ks[t] = k[base + t];
    vs[t] = v[base + t];
#endif
    __syncthreads();

    const float qi = q[base + t] * 0.088388347648318447f;  // 1/sqrt(128)
    float mx = -3.4e38f;
    for (int j = 0; j < DHEAD; ++j) mx = fmaxf(mx, qi * ks[j]);
    float sum = 0.f, acc = 0.f;
    for (int j = 0; j < DHEAD; ++j) {
        const float e = __expf(qi * ks[j] - mx);
        sum += e;
        acc += e * vs[j];
    }
    o[base + t] = (_Float16)(acc / sum);
}

// ------------------------------------------------------------------
// WMMA GEMM: C[M,N] = A(f16,[M,K]) * W(f32->f16,[K,N]) + epilogue
// Block: 256 thr = 8 waves, tile 128(M) x 64(N), K-step 32.
// Ping-pong LDS for the weight tile; W prefetched into registers one
// K-step ahead; single barrier per K-step; staging commits with one
// ds_store_b128 per thread (column-strip mapping).
// ------------------------------------------------------------------
enum { EPI_BIAS_F32 = 0, EPI_RESID_F32 = 1, EPI_RELU_F16 = 2 };

template <bool HEADSLICED>
__device__ __forceinline__ size_t wofs(int kk, int n, int N, int K)
{
    // HEADSLICED: W is [H, K, DHEAD]; logical col n = h*DHEAD + e
    if (HEADSLICED)
        return (size_t)(n >> 7) * (size_t)K * DHEAD + (size_t)kk * DHEAD + (n & 127);
    return (size_t)kk * N + n;
}

// A fragment, 16x32 f16 (ISA 7.12.2): lane m = l&15; half g = l>>4;
// VGPR j holds K pair { (j>>2)*16 + g*8 + (j&3)*2 , +1 }
__device__ __forceinline__ v16h load_A_frag(const _Float16* __restrict__ A,
                                            int lda, int lane)
{
    const int m = lane & 15;
    const int g = lane >> 4;
    const _Float16* p = A + (size_t)m * lda;
    v16h a;
#pragma unroll
    for (int j = 0; j < 8; ++j) {
        const int kk = ((j >> 2) << 4) + (g << 3) + ((j & 3) << 1);
        a[2 * j]     = p[kk];
        a[2 * j + 1] = p[kk + 1];
    }
    return a;
}

// B fragment, 32x16 f16 (per SWMMAC 64x16 pattern scaled to K=32):
// lane col n = l&15; half g = l>>4 (same K order as the A fragment),
// VGPR j holds K pair {(j>>2)*16 + g*8 + (j&3)*2, +1}.  Bs is stored K-contiguous per col.
__device__ __forceinline__ v16h load_B_frag(const _Float16* __restrict__ Bs,
                                            int col0, int ldb, int lane)
{
    const int n = lane & 15;
    const int g = lane >> 4;
    const _Float16* p = Bs + (size_t)(col0 + n) * ldb + (g << 3);
    v16h b;
#pragma unroll
    for (int j = 0; j < 8; ++j) {
        b[2 * j]     = p[((j >> 2) << 3) + 2 * j];
        b[2 * j + 1] = p[((j >> 2) << 3) + 2 * j + 1];
    }
    return b;
}

template <int EPI, bool HEADSLICED>
__global__ void __launch_bounds__(256)
gemm_kernel(const _Float16* __restrict__ A, const float* __restrict__ W,
            const float* __restrict__ bias, const float* __restrict__ resid,
            float* __restrict__ outF, _Float16* __restrict__ outH,
            int M, int N, int K)
{
    constexpr int BK  = 32;
    constexpr int BN  = 64;
    constexpr int LDB = 40;   // halfs per LDS column: 80 B keeps b128 stores
                              // 16-B aligned; 20-bank stride, conflict-free
    __shared__ __align__(16) _Float16 Bs[2 * BN * LDB];   // ping-pong [col][k]

    const int tid  = threadIdx.x;
    const int wave = tid >> 5;                    // 8 waves: split M
    const int lane = tid & 31;
    const int n0   = blockIdx.x * BN;
    const int m0   = blockIdx.y * 128 + wave * 16;

    // cooperative W-tile mapping: thread -> (one column, 8 consecutive k)
    const int wc = tid & 63;                      // column 0..63
    const int wk = (tid >> 6) * 8;                // k base: 0,8,16,24
    const size_t wstride = HEADSLICED ? DHEAD : (size_t)N;  // per-k stride

    v8f acc[4] = {};

    // preload W tile 0 into registers and A fragment 0
    float wreg[8];
    {
        const size_t wb = wofs<HEADSLICED>(wk, n0 + wc, N, K);
#pragma unroll
        for (int i = 0; i < 8; ++i) wreg[i] = W[wb + i * wstride];
    }
    v16h a_cur = load_A_frag(A + (size_t)m0 * K, K, lane);

    int buf = 0;
    for (int k0 = 0; k0 < K; k0 += BK) {
        // commit current W tile (fp32 -> f16) with a single b128 LDS store
        _Float16* bs = &Bs[buf * (BN * LDB)];
        {
            v8h pack;
#pragma unroll
            for (int i = 0; i < 8; ++i) pack[i] = (_Float16)wreg[i];
            *(v8h*)(&bs[wc * LDB + wk]) = pack;
        }

        // issue next tile's global loads now: they fly under the WMMAs
        if (k0 + BK < K) {
            const size_t wb = wofs<HEADSLICED>(k0 + BK + wk, n0 + wc, N, K);
#pragma unroll
            for (int i = 0; i < 8; ++i) wreg[i] = W[wb + i * wstride];
        }
        if (k0 + 2 * BK < K)
            __builtin_prefetch(&W[wofs<HEADSLICED>(k0 + 2 * BK + wk, n0 + wc, N, K)], 0, 0);

        __syncthreads();   // tile visible; prior-step readers already drained

        v16h a_next = a_cur;
        if (k0 + BK < K)
            a_next = load_A_frag(A + (size_t)m0 * K + k0 + BK, K, lane);

#pragma unroll
        for (int t = 0; t < 4; ++t) {
            const v16h bfrag = load_B_frag(bs, t * 16, LDB, lane);
            acc[t] = __builtin_amdgcn_wmma_f32_16x16x32_f16(
                false, a_cur, false, bfrag, (short)0, acc[t], false, false);
        }
        a_cur = a_next;
        buf ^= 1;
    }

    // C/D layout: VGPR r -> M = r + (lane>>4)*8, N = lane&15
    const int g  = lane >> 4;
    const int nn = lane & 15;
#pragma unroll
    for (int t = 0; t < 4; ++t) {
        const int col = n0 + t * 16 + nn;
        const float bv = bias[col];
#pragma unroll
        for (int r = 0; r < 8; ++r) {
            const int row = m0 + g * 8 + r;
            float val = acc[t][r] + bv;
            if (EPI == EPI_RESID_F32) val += resid[(size_t)row * N + col];
            if (EPI == EPI_RELU_F16) {
                outH[(size_t)row * N + col] = (_Float16)fmaxf(val, 0.f);
            } else {
                outF[(size_t)row * N + col] = val;
            }
        }
    }
}

// ------------------------------------------------------------------
// Launch
// ------------------------------------------------------------------
extern "C" void kernel_launch(void* const* d_in, const int* in_sizes, int n_in,
                              void* d_out, int out_size, void* d_ws, size_t ws_size,
                              hipStream_t stream)
{
    const float* x   = (const float*)d_in[0];
    const float* Wq  = (const float*)d_in[1];
    const float* bq  = (const float*)d_in[2];
    const float* Wk  = (const float*)d_in[3];
    const float* bk  = (const float*)d_in[4];
    const float* Wv  = (const float*)d_in[5];
    const float* bv  = (const float*)d_in[6];
    const float* Wo  = (const float*)d_in[7];
    const float* bo  = (const float*)d_in[8];
    const float* W1  = (const float*)d_in[9];
    const float* b1  = (const float*)d_in[10];
    const float* W2  = (const float*)d_in[11];
    const float* b2  = (const float*)d_in[12];
    const float* g1  = (const float*)d_in[13];
    const float* be1 = (const float*)d_in[14];
    const float* g2  = (const float*)d_in[15];
    const float* be2 = (const float*)d_in[16];
    (void)in_sizes; (void)n_in; (void)out_size; (void)ws_size;

    char* ws = (char*)d_ws;
    size_t off = 0;
    auto salloc = [&](size_t bytes) {
        void* p = ws + off;
        off += (bytes + 255) & ~(size_t)255;
        return p;
    };
    _Float16* h1  = (_Float16*)salloc((size_t)BATCH * D_MODEL * sizeof(_Float16));
    float*    qb  = (float*)   salloc((size_t)BATCH * D_MODEL * sizeof(float));
    float*    kb  = (float*)   salloc((size_t)BATCH * D_MODEL * sizeof(float));
    float*    vb  = (float*)   salloc((size_t)BATCH * D_MODEL * sizeof(float));
    _Float16* o16 = (_Float16*)salloc((size_t)BATCH * D_MODEL * sizeof(_Float16));
    float*    x1  = (float*)   salloc((size_t)BATCH * D_MODEL * sizeof(float));
    _Float16* h2  = (_Float16*)salloc((size_t)BATCH * D_MODEL * sizeof(_Float16));
    _Float16* a1  = (_Float16*)salloc((size_t)BATCH * DFF     * sizeof(_Float16));

    const dim3 blk(256);

    // LN1: x -> h1 (f16)
    ln_f16_kernel<<<BATCH, blk, 0, stream>>>(x, g1, be1, h1);

    // QKV projections (head-sliced weights [H,D,DH])
    const dim3 gD(D_MODEL / 64, BATCH / 128);
    gemm_kernel<EPI_BIAS_F32, true><<<gD, blk, 0, stream>>>(
        h1, Wq, bq, nullptr, qb, nullptr, BATCH, D_MODEL, D_MODEL);
    gemm_kernel<EPI_BIAS_F32, true><<<gD, blk, 0, stream>>>(
        h1, Wk, bk, nullptr, kb, nullptr, BATCH, D_MODEL, D_MODEL);
    gemm_kernel<EPI_BIAS_F32, true><<<gD, blk, 0, stream>>>(
        h1, Wv, bv, nullptr, vb, nullptr, BATCH, D_MODEL, D_MODEL);

    // per-(b,h) outer-product softmax attention -> o (f16)
    attn_kernel<<<BATCH * NHEAD, dim3(128), 0, stream>>>(qb, kb, vb, o16);

    // x1 = x + o @ Wo + bo
    gemm_kernel<EPI_RESID_F32, false><<<gD, blk, 0, stream>>>(
        o16, Wo, bo, x, x1, nullptr, BATCH, D_MODEL, D_MODEL);

    // LN2: x1 -> h2 (f16)
    ln_f16_kernel<<<BATCH, blk, 0, stream>>>(x1, g2, be2, h2);

    // a1 = relu(h2 @ W1 + b1)  (f16)
    const dim3 gFF(DFF / 64, BATCH / 128);
    gemm_kernel<EPI_RELU_F16, false><<<gFF, blk, 0, stream>>>(
        h2, W1, b1, nullptr, nullptr, a1, BATCH, DFF, D_MODEL);

    // out = x1 + a1 @ W2 + b2
    gemm_kernel<EPI_RESID_F32, false><<<gD, blk, 0, stream>>>(
        a1, W2, b2, x1, (float*)d_out, nullptr, BATCH, D_MODEL, DFF);
}
